// _Ltx2Attention_26319559590204
// MI455X (gfx1250) — compile-verified, matrix operand order checked by reading, not run on hardware
//
#include <hip/hip_runtime.h>
#include <math.h>

#define B_  2
#define T_  2048
#define D_  1024
#define H_  16
#define DH_ 64
#define M_  (B_*T_)   // 4096 rows (b,t)
#define K_  D_
#define N_  D_
#define AW  4         // waves per attention block

typedef __attribute__((ext_vector_type(16))) _Float16 v16h;
typedef __attribute__((ext_vector_type(8)))  _Float16 h8;
typedef __attribute__((ext_vector_type(8)))  float    v8f;

union HF { v16h v; h8 h[2]; };

#define LDS_FENCE() asm volatile("s_wait_dscnt 0" ::: "memory")

__device__ __forceinline__ v8f wmma_f16(const HF& a, const HF& b, v8f c) {
  return __builtin_amdgcn_wmma_f32_16x16x32_f16(false, a.v, false, b.v, (short)0, c, false, false);
}

// ---------------- fp32 -> fp16 convert ----------------
__global__ void k_cvt_f16(const float* __restrict__ x, _Float16* __restrict__ y, int n) {
  int i = (blockIdx.x * blockDim.x + threadIdx.x) * 4;
  if (i + 3 < n) {
    float4 v = *(const float4*)(x + i);
    y[i+0] = (_Float16)v.x; y[i+1] = (_Float16)v.y;
    y[i+2] = (_Float16)v.z; y[i+3] = (_Float16)v.w;
  }
}

// ---------------- tiled transpose + convert: W[K][N] -> Wt[N][K] (f16) ----------------
__global__ void k_transpose_cvt(const float* __restrict__ w, _Float16* __restrict__ wt) {
  __shared__ float tile[32][33];
  int k0 = blockIdx.x * 32, n0 = blockIdx.y * 32;
  int tx = threadIdx.x, ty = threadIdx.y;
#pragma unroll
  for (int i = 0; i < 4; ++i)
    tile[ty + i*8][tx] = w[(size_t)(k0 + ty + i*8)*N_ + n0 + tx];
  __syncthreads();
#pragma unroll
  for (int i = 0; i < 4; ++i)
    wt[(size_t)(n0 + ty + i*8)*K_ + k0 + tx] = (_Float16)tile[tx][ty + i*8];
}

// ---------------- WMMA GEMM: C[M][N] = A[M][K] * Bt[N][K]^T + bias ----------------
// Register-blocked: each wave computes a 16x64 strip (4 N-tiles) so one A
// fragment feeds 4 WMMAs -> 2.5 b128 loads per WMMA instead of 4, and 4
// independent accumulator chains hide global_load latency.
// EPI==0: store fp32 row-major to outp.  EPI==1: store f16 to Vt[b][h][dh][t].
template<int EPI>
__global__ void k_gemm(const _Float16* __restrict__ A, const _Float16* __restrict__ Bt,
                       const float* __restrict__ bias, void* __restrict__ outp) {
  int lane = threadIdx.x & 31;
  int wv   = threadIdx.x >> 5;
  int m0 = (blockIdx.y * 8 + wv) * 16;
  int n0 = blockIdx.x * 64;
  int r16 = lane & 15;
  int cg  = lane >> 4;
  int klo = cg * 8;
  const _Float16* ap = A  + (size_t)(m0 + r16) * K_ + klo;
  const _Float16* bp = Bt + (size_t)(n0 + r16) * K_ + klo;
  v8f acc[4] = {};
#pragma unroll 2
  for (int k0 = 0; k0 < K_; k0 += 32) {
    HF a;
    a.h[0] = *(const h8*)(ap);
    a.h[1] = *(const h8*)(ap + 16);
    __builtin_prefetch(ap + 64, 0, 1);
#pragma unroll
    for (int nt = 0; nt < 4; ++nt) {
      const _Float16* bpn = bp + (size_t)nt * 16 * K_;
      HF b;
      b.h[0] = *(const h8*)(bpn);
      b.h[1] = *(const h8*)(bpn + 16);
      acc[nt] = wmma_f16(a, b, acc[nt]);
    }
    __builtin_prefetch(bp + 64, 0, 1);
    ap += 32; bp += 32;
  }
#pragma unroll
  for (int nt = 0; nt < 4; ++nt) {
    int n = n0 + nt*16 + r16;
    float bs = bias[n];
#pragma unroll
    for (int r = 0; r < 8; ++r) {
      int m = m0 + r + 8*cg;
      float val = acc[nt][r] + bs;
      if (EPI == 0) {
        ((float*)outp)[(size_t)m * N_ + n] = val;
      } else {
        int b_ = m >> 11, t = m & (T_-1);
        int h  = n >> 6,  dh = n & (DH_-1);
        ((_Float16*)outp)[(size_t)((b_*H_ + h)*DH_ + dh)*T_ + t] = (_Float16)val;
      }
    }
  }
}

// ---------------- RMSNorm + interleaved RoPE, fp32 -> f16 head-major [B,H,T,DH] ----------------
__global__ void k_rmsnorm_rope(const float* __restrict__ Qf, const float* __restrict__ Kf,
                               const float* __restrict__ gq, const float* __restrict__ gk,
                               const float* __restrict__ cosp, const float* __restrict__ sinp,
                               _Float16* __restrict__ Qh, _Float16* __restrict__ Kh) {
  int row = blockIdx.x;                 // b*T + t
  const float* src = blockIdx.y ? Kf : Qf;
  const float* g   = blockIdx.y ? gk : gq;
  _Float16* dst    = blockIdx.y ? Kh : Qh;
  int t = row & (T_-1), b = row >> 11;
  int tid = threadIdx.x;
  int d0  = tid * 4;
  float4 x = *(const float4*)(src + (size_t)row*D_ + d0);
  __shared__ float red[256];
  red[tid] = x.x*x.x + x.y*x.y + x.z*x.z + x.w*x.w;
  __syncthreads();
  for (int s = 128; s > 0; s >>= 1) {
    if (tid < s) red[tid] += red[tid + s];
    __syncthreads();
  }
  float rs = rsqrtf(red[0] * (1.0f / D_) + 1e-6f);
  float y0 = x.x * rs * g[d0+0];
  float y1 = x.y * rs * g[d0+1];
  float y2 = x.z * rs * g[d0+2];
  float y3 = x.w * rs * g[d0+3];
  const float* cp = cosp + (size_t)t*D_ + d0;
  const float* sp = sinp + (size_t)t*D_ + d0;
  float o0 = y0*cp[0] - y1*sp[0];
  float o1 = y1*cp[1] + y0*sp[1];
  float o2 = y2*cp[2] - y3*sp[2];
  float o3 = y3*cp[3] + y2*sp[3];
  int h = d0 >> 6, dh = d0 & (DH_-1);
  _Float16* q = dst + ((size_t)(b*H_ + h)*T_ + t)*DH_ + dh;
  q[0] = (_Float16)o0; q[1] = (_Float16)o1; q[2] = (_Float16)o2; q[3] = (_Float16)o3;
}

// ---------------- Flash attention: one wave = 16 queries, online softmax over 32-key chunks ----------------
__global__ void k_attn(const _Float16* __restrict__ Qh, const _Float16* __restrict__ Kh,
                       const _Float16* __restrict__ Vt, _Float16* __restrict__ Oh) {
  __shared__ alignas(16) float    sS[AW][16*32];
  __shared__ alignas(16) _Float16 sP[AW][16*32];
  __shared__ float sAlpha[AW][16];
  __shared__ float sInv[AW][16];

  int lane = threadIdx.x & 31;
  int w    = threadIdx.x >> 5;
  int g    = blockIdx.x * AW + w;
  int qt   = g & (T_/16 - 1);   // 128 q-tiles per (b,h)
  int bh   = g >> 7;
  int row  = lane & 15;
  int cg   = lane >> 4;
  int klo  = cg * 8;

  // Q fragments (A-operand, 16x64 split as two 16x32) held in registers for whole T loop
  const _Float16* qr = Qh + ((size_t)bh*T_ + qt*16 + row) * DH_;
  HF aq0, aq1;
  aq0.h[0] = *(const h8*)(qr + klo);       aq0.h[1] = *(const h8*)(qr + klo + 16);
  aq1.h[0] = *(const h8*)(qr + 32 + klo);  aq1.h[1] = *(const h8*)(qr + 32 + klo + 16);

  v8f o[4] = {};
  float mrow = -INFINITY, lrow = 0.0f;
  const _Float16* kbase = Kh + (size_t)bh*T_*DH_;
  const _Float16* vbase = Vt + (size_t)bh*DH_*T_;

  for (int kc = 0; kc < T_/32; ++kc) {
    int key0 = kc * 32;
    // S = Q K^T for 32 keys (two 16-key n-tiles, each 2 WMMAs over DH=64)
#pragma unroll
    for (int nt = 0; nt < 2; ++nt) {
      const _Float16* kr = kbase + (size_t)(key0 + nt*16 + row) * DH_;
      HF b0, b1;
      b0.h[0] = *(const h8*)(kr + klo);       b0.h[1] = *(const h8*)(kr + klo + 16);
      b1.h[0] = *(const h8*)(kr + 32 + klo);  b1.h[1] = *(const h8*)(kr + 32 + klo + 16);
      v8f s = {};
      s = wmma_f16(aq0, b0, s);
      s = wmma_f16(aq1, b1, s);
#pragma unroll
      for (int r = 0; r < 8; ++r)
        sS[w][(r + 8*cg)*32 + nt*16 + row] = s[r] * 0.125f;   // 1/sqrt(DH)
    }
    LDS_FENCE();

    // online softmax: this lane owns row `row`, columns [cg*16, cg*16+16)
    float sv[16];
    const float* sr = &sS[w][row*32 + cg*16];
#pragma unroll
    for (int j = 0; j < 16; ++j) sv[j] = sr[j];
    float cmax = sv[0];
#pragma unroll
    for (int j = 1; j < 16; ++j) cmax = fmaxf(cmax, sv[j]);
    cmax = fmaxf(cmax, __shfl_xor(cmax, 16, 32));
    float mnew  = fmaxf(mrow, cmax);
    float alpha = __expf(mrow - mnew);
    float psum  = 0.0f;
    _Float16* pr = &sP[w][row*32 + cg*16];
#pragma unroll
    for (int j = 0; j < 16; ++j) {
      float p = __expf(sv[j] - mnew);
      psum += p;
      pr[j] = (_Float16)p;
    }
    psum += __shfl_xor(psum, 16, 32);
    lrow = lrow * alpha + psum;
    mrow = mnew;
    if (cg == 0) sAlpha[w][row] = alpha;
    LDS_FENCE();

    // rescale accumulators by per-row alpha (C-layout: rows r+8*cg)
#pragma unroll
    for (int r = 0; r < 8; ++r) {
      float al = sAlpha[w][r + 8*cg];
      o[0][r] *= al; o[1][r] *= al; o[2][r] *= al; o[3][r] *= al;
    }
    // P (A-operand 16x32) from LDS
    HF p;
    p.h[0] = *(const h8*)(&sP[w][row*32 + klo]);
    p.h[1] = *(const h8*)(&sP[w][row*32 + klo + 16]);
    // O += P @ V  (4 dh-tiles of 16, V transposed so keys are contiguous)
#pragma unroll
    for (int nt = 0; nt < 4; ++nt) {
      const _Float16* vr = vbase + (size_t)(nt*16 + row) * T_ + key0;
      HF bv;
      bv.h[0] = *(const h8*)(vr + klo);
      bv.h[1] = *(const h8*)(vr + klo + 16);
      o[nt] = wmma_f16(p, bv, o[nt]);
    }
  }

  if (cg == 0) sInv[w][row] = 1.0f / lrow;
  LDS_FENCE();
  int b = bh >> 4, h = bh & (H_-1);
#pragma unroll
  for (int r = 0; r < 8; ++r) {
    float inv = sInv[w][r + 8*cg];
    int m = r + 8*cg;
    size_t t = (size_t)qt*16 + m;
    _Float16* od = Oh + ((size_t)b*T_ + t)*D_ + h*DH_ + row;
    od[0]  = (_Float16)(o[0][r] * inv);
    od[16] = (_Float16)(o[1][r] * inv);
    od[32] = (_Float16)(o[2][r] * inv);
    od[48] = (_Float16)(o[3][r] * inv);
  }
}

// ---------------- host launch ----------------
extern "C" void kernel_launch(void* const* d_in, const int* in_sizes, int n_in,
                              void* d_out, int out_size, void* d_ws, size_t ws_size,
                              hipStream_t stream) {
  const float* hs   = (const float*)d_in[0];
  const float* cosp = (const float*)d_in[1];
  const float* sinp = (const float*)d_in[2];
  const float* wq   = (const float*)d_in[3];
  const float* bq   = (const float*)d_in[4];
  const float* wk   = (const float*)d_in[5];
  const float* bk   = (const float*)d_in[6];
  const float* wv   = (const float*)d_in[7];
  const float* bv   = (const float*)d_in[8];
  const float* gq   = (const float*)d_in[9];
  const float* gk   = (const float*)d_in[10];
  const float* wo   = (const float*)d_in[11];
  const float* bo   = (const float*)d_in[12];

  char* p = (char*)d_ws;
  auto take = [&](size_t bytes) -> char* {
    char* r = p; p += (bytes + 255) & ~(size_t)255; return r;
  };
  _Float16* Xh  = (_Float16*)take((size_t)M_*K_*2);
  _Float16* WqT = (_Float16*)take((size_t)K_*N_*2);
  _Float16* WkT = (_Float16*)take((size_t)K_*N_*2);
  _Float16* WvT = (_Float16*)take((size_t)K_*N_*2);
  _Float16* WoT = (_Float16*)take((size_t)K_*N_*2);
  float*    Qf  = (float*)   take((size_t)M_*N_*4);
  float*    Kf  = (float*)   take((size_t)M_*N_*4);
  _Float16* Qh  = (_Float16*)take((size_t)M_*N_*2);
  _Float16* Kh  = (_Float16*)take((size_t)M_*N_*2);
  _Float16* Vt  = (_Float16*)take((size_t)B_*H_*DH_*T_*2);
  _Float16* Oh  = (_Float16*)take((size_t)M_*N_*2);

  // 1) fp16 staging
  k_cvt_f16<<<(M_*K_)/1024, 256, 0, stream>>>(hs, Xh, M_*K_);
  dim3 tb(32, 8), tg(K_/32, N_/32);
  k_transpose_cvt<<<tg, tb, 0, stream>>>(wq, WqT);
  k_transpose_cvt<<<tg, tb, 0, stream>>>(wk, WkT);
  k_transpose_cvt<<<tg, tb, 0, stream>>>(wv, WvT);
  k_transpose_cvt<<<tg, tb, 0, stream>>>(wo, WoT);

  // 2) QKV projections via WMMA (16x64 strip per wave)
  dim3 gg(N_/64, M_/16/8);
  k_gemm<0><<<gg, 256, 0, stream>>>(Xh, WqT, bq, Qf);
  k_gemm<0><<<gg, 256, 0, stream>>>(Xh, WkT, bk, Kf);
  k_gemm<1><<<gg, 256, 0, stream>>>(Xh, WvT, bv, Vt);   // f16 [B,H,DH,T]

  // 3) RMSNorm + RoPE -> f16 [B,H,T,DH]
  dim3 rg(B_*T_, 2);
  k_rmsnorm_rope<<<rg, 256, 0, stream>>>(Qf, Kf, gq, gk, cosp, sinp, Qh, Kh);

  // 4) flash attention -> Oh f16 [B,T,D]
  k_attn<<<(B_*H_*T_/16)/AW, AW*32, 0, stream>>>(Qh, Kh, Vt, Oh);

  // 5) output projection -> fp32 d_out
  k_gemm<0><<<gg, 256, 0, stream>>>(Oh, WoT, bo, (float*)d_out);
}
